// CornerPool_90847148245475
// MI455X (gfx1250) — matrix operand order repaired and checked by reading, not compiled, not run on hardware
//
#include <hip/hip_runtime.h>
#include <hip/hip_bf16.h>

typedef __attribute__((ext_vector_type(16))) __bf16    v16bf;
typedef __attribute__((ext_vector_type(8)))  float     v8f;
typedef __attribute__((ext_vector_type(4)))  unsigned  v4u;

struct Frag32B { v4u lo, hi; };                  // 32 bytes -> bit_cast to v16bf

#define HW    16384
#define IMG_H 128
#define IMG_W 128
#define NB    8
#define SPITCH 40                                 // LDS column pitch (bf16 elems); 80B, 16B-aligned

__device__ __forceinline__ __bf16 f2bf(float f) {
  unsigned u = __builtin_bit_cast(unsigned, f);
  u += 0x7FFFu + ((u >> 16) & 1u);               // round-to-nearest-even
  unsigned short s = (unsigned short)(u >> 16);
  return __builtin_bit_cast(__bf16, s);
}

// pack two floats into a dword of two bf16 (lo = a, hi = b)
__device__ __forceinline__ unsigned pk2bf(float a, float b) {
  unsigned ua = __builtin_bit_cast(unsigned, a);
  ua += 0x7FFFu + ((ua >> 16) & 1u);
  unsigned ub = __builtin_bit_cast(unsigned, b);
  ub += 0x7FFFu + ((ub >> 16) & 1u);
  return (ua >> 16) | (ub & 0xFFFF0000u);
}

// weight prep: f32 OIHW [co][ci][rr] -> bf16 [co][rr][ci]  (k-order = (khw, ci))
__global__ void wt_prep_bf16(const float* __restrict__ src, __bf16* __restrict__ dst,
                             int cin, int rr, int n) {
  int i = blockIdx.x * blockDim.x + threadIdx.x;
  if (i >= n) return;
  int ci  = i % cin;                              // cin is a power of two
  int t   = i / cin;
  int khw = t % rr;
  int co  = t / rr;
  dst[i] = f2bf(src[((size_t)co * cin + ci) * rr + khw]);
}

// Implicit-GEMM conv (R=1 or 3, pad=R/2) + fused inference-BN (+residual)(+relu).
// K-order = (kh, kw, ci): tap decomposition / pad checks hoisted out of the
// 32-wide K inner loop. One wave -> (16*CW) co x 64 pixels: CW A fragments,
// 4 B fragments, CW*4 back-to-back WMMAs per K-step. 4 waves/block share the
// LDS activation tile (32 K x 64 pix bf16).
template<int CIN, int COUT, int R, bool RELU, bool RESADD>
__global__ __launch_bounds__(128, 2)
void conv_bn_wmma(const float* __restrict__ x,
                  const __bf16* __restrict__ wt,   // [COUT][R*R][CIN] bf16
                  const float* __restrict__ bn,    // [4][COUT] gamma,beta,mean,var
                  const float* __restrict__ res,
                  float* __restrict__ out)
{
  constexpr int K  = CIN * R * R;
  constexpr int CW = (COUT >= 128) ? 2 : 1;        // co tiles per wave
  __shared__ __bf16 sb[64 * SPITCH];               // [pix][k], 5.1 KB

  const int tid  = threadIdx.x;
  const int lane = tid & 31;
  const int wave = tid >> 5;
  const int half = lane >> 4;
  const int nl   = lane & 15;                      // B/D column base; A row

  const int p0 = blockIdx.x << 6;                  // 64 consecutive pixels, one image row
  const int b  = p0 >> 14;
  const int rp = p0 & (HW - 1);
  const int h0 = rp >> 7;
  const int w0 = rp & (IMG_W - 1);

  const int co_base = (blockIdx.y * 4 + wave) * (16 * CW);

  // staging: each thread owns pixel column sn and 16 contiguous K rows
  const int sn = tid & 63;
  const int k0 = (tid >> 6) << 4;                  // 0 or 16

  v8f acc[CW][4] = {};

  #pragma unroll 1
  for (int khw = 0; khw < R * R; ++khw) {
    // per-tap geometry (uniform over the K inner loop)
    bool valid;
    size_t xoff;
    if (R == 3) {
      const int kh = khw / 3;
      const int kw = khw - kh * 3;
      const int hh  = h0 + kh - 1;
      const int wwt = w0 + sn + kw - 1;
      valid = ((unsigned)hh < IMG_H) && ((unsigned)wwt < IMG_W);
      xoff  = (((size_t)b * CIN) << 14) + (size_t)(valid ? hh * IMG_W + wwt : 0);
    } else {
      valid = true;
      xoff  = (((size_t)b * CIN) << 14) + (size_t)(h0 * IMG_W + w0 + sn);
    }
    const float* xp = x + xoff;

    #pragma unroll 1
    for (int ci0 = 0; ci0 < CIN; ci0 += 32) {
      __syncthreads();
      // cooperative stage: 32 ci x 64 pixels, packed bf16 pairs (2x ds_store_b128)
      #pragma unroll
      for (int i = 0; i < 8; ++i) {
        int kl = k0 + 2 * i;
        float v0 = 0.f, v1 = 0.f;
        if (valid) {
          v0 = xp[(size_t)(ci0 + kl)     << 14];
          v1 = xp[(size_t)(ci0 + kl + 1) << 14];
        }
        *(unsigned*)(sb + sn * SPITCH + kl) = pk2bf(v0, v1);
      }
      __syncthreads();

      // A fragments: 16x32 bf16 weights; lane half selects K 0-7/16-23 vs 8-15/24-31
      v16bf a[CW];
      #pragma unroll
      for (int c = 0; c < CW; ++c) {
        const __bf16* ap = wt + (size_t)(co_base + c * 16 + nl) * K + (khw * CIN + ci0);
        Frag32B araw;
        araw.lo = *(const v4u*)(ap + (half << 3));
        araw.hi = *(const v4u*)(ap + 16 + (half << 3));
        a[c] = __builtin_bit_cast(v16bf, araw);
      }

      // 4 B fragments loaded up-front into distinct registers
      v16bf bm[4];
      #pragma unroll
      for (int t = 0; t < 4; ++t) {
        const __bf16* bp = sb + ((t << 4) + nl) * SPITCH + (half << 3);
        Frag32B braw;
        braw.lo = *(const v4u*)bp;
        braw.hi = *(const v4u*)(bp + 16);
        bm[t] = __builtin_bit_cast(v16bf, braw);
      }
      // keep loads above, WMMAs below: one wait, then back-to-back matrix ops
      __builtin_amdgcn_sched_barrier(0);
      #pragma unroll
      for (int t = 0; t < 4; ++t)
        #pragma unroll
        for (int c = 0; c < CW; ++c)
          acc[c][t] = __builtin_amdgcn_wmma_f32_16x16x32_bf16(
              false, a[c], false, bm[t], (short)0, acc[c][t], false, false);
    }
  }

  // epilogue: BN scale/bias (+res)(+relu); D VGPR r <-> rows r (lanes 0-15) / r+8 (16-31)
  #pragma unroll
  for (int c = 0; c < CW; ++c) {
    #pragma unroll
    for (int r = 0; r < 8; ++r) {
      int co = co_base + c * 16 + r + 8 * half;
      float g  = bn[co];
      float be = bn[COUT + co];
      float mn = bn[2 * COUT + co];
      float va = bn[3 * COUT + co];
      float sc = g * rsqrtf(va + 1e-5f);
      float bi = be - mn * sc;
      size_t orow = (((size_t)b * COUT + co) << 14) + (size_t)(h0 * IMG_W + w0 + nl);
      #pragma unroll
      for (int t = 0; t < 4; ++t) {
        float val = acc[c][t][r] * sc + bi;
        size_t oidx = orow + (t << 4);
        if (RESADD) val += res[oidx];
        if (RELU)   val = fmaxf(val, 0.f);
        out[oidx] = val;
      }
    }
  }
}

// reverse cumulative max over H, in place; one thread per (b,c,w) line
__global__ void cummax_rev_h(float* __restrict__ t, int nlines) {
  int i = blockIdx.x * blockDim.x + threadIdx.x;
  if (i >= nlines) return;
  int wcol = i & (IMG_W - 1);
  size_t bc = (size_t)(i >> 7);
  float* base = t + (bc << 14) + wcol;
  float m = -3.402823466e38f;
  for (int h = IMG_H - 1; h >= 0; --h) {
    float v = base[(size_t)h << 7];
    m = fmaxf(m, v);
    base[(size_t)h << 7] = m;
  }
}

// reverse cumulative max over W, in place; one thread per (b,c,h) row
__global__ void cummax_rev_w(float* __restrict__ t, int nlines) {
  int i = blockIdx.x * blockDim.x + threadIdx.x;
  if (i >= nlines) return;
  float* base = t + ((size_t)i << 7);
  float m = -3.402823466e38f;
  for (int w = IMG_W - 1; w >= 0; --w) {
    float v = base[w];
    m = fmaxf(m, v);
    base[w] = m;
  }
}

__global__ void add_inplace(float* __restrict__ a, const float* __restrict__ b, int n) {
  int i = blockIdx.x * blockDim.x + threadIdx.x;
  if (i < n) a[i] += b[i];
}

extern "C" void kernel_launch(void* const* d_in, const int* in_sizes, int n_in,
                              void* d_out, int out_size, void* d_ws, size_t ws_size,
                              hipStream_t stream)
{
  const float* x       = (const float*)d_in[0];
  const float* w_res   = (const float*)d_in[1];
  const float* bn_res  = (const float*)d_in[2];
  const float* w_vpre  = (const float*)d_in[3];
  const float* bn_vpre = (const float*)d_in[4];
  const float* w_hpre  = (const float*)d_in[5];
  const float* bn_hpre = (const float*)d_in[6];
  const float* w_add   = (const float*)d_in[7];
  const float* bn_add  = (const float*)d_in[8];
  const float* w_post  = (const float*)d_in[9];
  const float* bn_post = (const float*)d_in[10];
  float* out = (float*)d_out;

  char* ws = (char*)d_ws;
  size_t off = 0;
  auto alloc = [&](size_t bytes) {
    void* p = ws + off;
    off = (off + bytes + 255) & ~(size_t)255;
    return p;
  };
  float*  res     = (float*)alloc(sizeof(float)  * (size_t)NB * 256 * HW);
  float*  p1      = (float*)alloc(sizeof(float)  * (size_t)NB * 64  * HW);
  float*  p2      = (float*)alloc(sizeof(float)  * (size_t)NB * 64  * HW);
  float*  mid     = (float*)alloc(sizeof(float)  * (size_t)NB * 256 * HW);
  __bf16* wb_res  = (__bf16*)alloc(sizeof(__bf16) * 256 * 256);
  __bf16* wb_vpre = (__bf16*)alloc(sizeof(__bf16) * 64 * 256 * 9);
  __bf16* wb_hpre = (__bf16*)alloc(sizeof(__bf16) * 64 * 256 * 9);
  __bf16* wb_add  = (__bf16*)alloc(sizeof(__bf16) * 256 * 64 * 9);
  __bf16* wb_post = (__bf16*)alloc(sizeof(__bf16) * 256 * 256 * 9);

  auto prep = [&](const float* s, __bf16* d, int cin, int rr, int cout) {
    int n = cout * cin * rr;
    wt_prep_bf16<<<(n + 255) / 256, 256, 0, stream>>>(s, d, cin, rr, n);
  };
  prep(w_res,  wb_res,  256, 1, 256);
  prep(w_vpre, wb_vpre, 256, 9, 64);
  prep(w_hpre, wb_hpre, 256, 9, 64);
  prep(w_add,  wb_add,  64,  9, 256);
  prep(w_post, wb_post, 256, 9, 256);

  const int NPIX = NB * HW;                         // 131072 output pixels
  dim3 blk(128);

  // residual branch: 1x1 conv 256->256, BN, no relu  (CW=2 -> grid.y = 2)
  conv_bn_wmma<256, 256, 1, false, false>
      <<<dim3(NPIX / 64, 2), blk, 0, stream>>>(x, wb_res, bn_res, nullptr, res);
  // vertical / horizontal pre-convs: 3x3 conv 256->64, BN, relu (CW=1 -> grid.y = 1)
  conv_bn_wmma<256, 64, 3, true, false>
      <<<dim3(NPIX / 64, 1), blk, 0, stream>>>(x, wb_vpre, bn_vpre, nullptr, p1);
  conv_bn_wmma<256, 64, 3, true, false>
      <<<dim3(NPIX / 64, 1), blk, 0, stream>>>(x, wb_hpre, bn_hpre, nullptr, p2);
  // corner pools (reverse cummax), in place
  cummax_rev_h<<<(NB * 64 * IMG_W + 255) / 256, 256, 0, stream>>>(p1, NB * 64 * IMG_W);
  cummax_rev_w<<<(NB * 64 * IMG_H + 255) / 256, 256, 0, stream>>>(p2, NB * 64 * IMG_H);
  add_inplace<<<(NB * 64 * HW + 255) / 256, 256, 0, stream>>>(p1, p2, NB * 64 * HW);
  // merge: 3x3 conv 64->256, BN, + residual, relu (CW=2 -> grid.y = 2)
  conv_bn_wmma<64, 256, 3, true, true>
      <<<dim3(NPIX / 64, 2), blk, 0, stream>>>(p1, wb_add, bn_add, res, mid);
  // post: 3x3 conv 256->256, BN, relu -> final output (CW=2 -> grid.y = 2)
  conv_bn_wmma<256, 256, 3, true, false>
      <<<dim3(NPIX / 64, 2), blk, 0, stream>>>(mid, wb_post, bn_post, nullptr, out);

  (void)in_sizes; (void)n_in; (void)out_size; (void)ws_size;
}
